// Subgraph_emb_43997644981186
// MI455X (gfx1250) — matrix operand order repaired and checked by reading, not compiled, not run on hardware
//
#include <hip/hip_runtime.h>
#include <hip/hip_bf16.h>

typedef __bf16 bf16;
typedef __attribute__((ext_vector_type(16))) __bf16 v16bf;
typedef __attribute__((ext_vector_type(8)))  __bf16 v8bf;
typedef __attribute__((ext_vector_type(8)))  float  v8f;

#define DD 128

// ---------------------------------------------------------------- zero
__global__ void zero_f32_kernel(float* __restrict__ p, int n4) {
  int i = blockIdx.x * blockDim.x + threadIdx.x;
  if (i < n4) reinterpret_cast<float4*>(p)[i] = make_float4(0.f, 0.f, 0.f, 0.f);
}

// ------------------------------------------- pack W (f32 row-major KxN) into
// WMMA bf16 B-fragments: frag[(kt*8+nt)*32 + lane][i] = W[kt*32 + (lane<16?0:16) + i][nt*16 + lane%16]
// 3 matrices x 4 kt x 8 nt x 32 lanes x 16 elems = 49152 threads
__global__ void prep_wfrag_kernel(const float* __restrict__ W1,
                                  const float* __restrict__ W2,
                                  const float* __restrict__ Wp,
                                  bf16* __restrict__ Wf) {
  int tid  = blockIdx.x * blockDim.x + threadIdx.x;
  int mat  = tid >> 14;          // 0..2
  int rem  = tid & 16383;
  int i    = rem & 15;
  int lane = (rem >> 4) & 31;
  int nt   = (rem >> 9) & 7;
  int kt   = rem >> 12;
  const float* W = (mat == 0) ? W1 : ((mat == 1) ? W2 : Wp);
  int n = nt * 16 + (lane & 15);
  int k = kt * 32 + ((lane < 16) ? 0 : 16) + i;
  Wf[tid] = (bf16)W[k * DD + n];
}

// ---------------------------------------------------------------- SpMM
// one wave per edge; lane handles 4 dims (float4 gather + 4 f32 atomics).
// Edge indices/weight are wave-uniform: broadcast to SGPRs via readfirstlane
// so the gather/scatter become saddr+voffset forms.
__global__ void spmm_kernel(const float* __restrict__ x,
                            const float* __restrict__ ew,
                            const int* __restrict__ src,
                            const int* __restrict__ dst,
                            float* __restrict__ agg, int nE) {
  int wid  = (blockIdx.x * blockDim.x + threadIdx.x) >> 5;
  int lane = threadIdx.x & 31;
  if (wid >= nE) return;
  int   s = __builtin_amdgcn_readfirstlane(src[wid]);
  int   d = __builtin_amdgcn_readfirstlane(dst[wid]);
  float w = __int_as_float(__builtin_amdgcn_readfirstlane(__float_as_int(ew[wid])));
  const float4 v = *reinterpret_cast<const float4*>(x + (size_t)s * DD + lane * 4);
  float* o = agg + (size_t)d * DD + lane * 4;
  unsafeAtomicAdd(o + 0, v.x * w);
  unsafeAtomicAdd(o + 1, v.y * w);
  unsafeAtomicAdd(o + 2, v.z * w);
  unsafeAtomicAdd(o + 3, v.w * w);
}

// ---------------------------------------------------------------- GEMM
// out[row] = relu(A[row] @ W + bias) (+ addsrc[row] if HAS_ADD)
// block = 256 threads = 8 waves; block computes 16 rows x 128 cols.
// A tile staged in LDS as bf16 (shared by all 8 waves); wave w owns N-tile w.
// GUARD=false: nrows is a multiple of 16, no row bounds checks.
template <bool HAS_ADD, bool GUARD>
__global__ void gcn_gemm_kernel(const float* __restrict__ A,
                                const bf16* __restrict__ Bf,
                                const float* __restrict__ bias,
                                const float* __restrict__ addsrc,
                                float* __restrict__ out, int nrows) {
  __shared__ alignas(16) bf16 ldsA[16 * DD];   // 4 KB
  const int tid     = threadIdx.x;
  const int rowBase = blockIdx.x * 16;

  // cooperative load: 16x128 f32 tile -> bf16 LDS (8 elems/thread, contiguous)
  {
    int base = tid * 8;                    // 0..2047
    float4 v0 = make_float4(0.f, 0.f, 0.f, 0.f), v1 = v0;
    bool ok = true;
    if (GUARD) ok = (rowBase + (base >> 7)) < nrows;
    if (ok) {
      const float4* ap = reinterpret_cast<const float4*>(A + (size_t)rowBase * DD + base);
      v0 = ap[0];
      v1 = ap[1];
    }
    bf16* lp = ldsA + base;
    lp[0] = (bf16)v0.x; lp[1] = (bf16)v0.y; lp[2] = (bf16)v0.z; lp[3] = (bf16)v0.w;
    lp[4] = (bf16)v1.x; lp[5] = (bf16)v1.y; lp[6] = (bf16)v1.z; lp[7] = (bf16)v1.w;
  }
  __syncthreads();

  const int wave = tid >> 5;       // N-tile index 0..7
  const int lane = tid & 31;
  const int m    = lane & 15;
  const int hi   = lane >> 4;      // 0: lanes 0-15, 1: lanes 16-31

  v8f c = {};
#pragma unroll
  for (int kt = 0; kt < 4; ++kt) {
    // A frag (contiguous K order, matches packed B): elems 0-15 -> K=koff..koff+15
    const int koff = kt * 32 + hi * 16;
    v8bf a0 = *reinterpret_cast<const v8bf*>(&ldsA[m * DD + koff]);
    v8bf a1 = *reinterpret_cast<const v8bf*>(&ldsA[m * DD + koff + 8]);
    v16bf a = __builtin_shufflevector(a0, a1, 0, 1, 2, 3, 4, 5, 6, 7,
                                              8, 9, 10, 11, 12, 13, 14, 15);
    // B frag: contiguous 32B per lane, pre-packed
    v16bf b = *reinterpret_cast<const v16bf*>(Bf + (((kt * 8 + wave) * 32 + lane) << 4));
    c = __builtin_amdgcn_wmma_f32_16x16x32_bf16(false, a, false, b,
                                                (short)0, c, false, false);
  }

  // C/D layout: VGPR r -> M = r (lanes 0-15) / r+8 (lanes 16-31), N = lane%16
  const int col   = wave * 16 + (lane & 15);
  const float bcl = bias[col];
#pragma unroll
  for (int r = 0; r < 8; ++r) {
    int row = rowBase + r + hi * 8;
    if (!GUARD || row < nrows) {
      size_t idx = (size_t)row * DD + col;
      float v = c[r] + bcl;
      v = v > 0.f ? v : 0.f;
      if (HAS_ADD) v += addsrc[idx];
      out[idx] = v;
    }
  }
}

// ---------------------------------------------------------------- pooling
// contiguous equal-size subgraphs: start = target[b], count = size_subg[b]
__global__ void pool_kernel(const float* __restrict__ h,
                            const int* __restrict__ target,
                            const float* __restrict__ size_subg,
                            float* __restrict__ pooled) {
  int b   = blockIdx.x;
  int col = threadIdx.x;  // 128
  int start = __builtin_amdgcn_readfirstlane(target[b]);
  float cnt = size_subg[b];
  int n = (int)cnt;
  const float* hp = h + (size_t)start * DD + col;
  float s = 0.f;
  for (int i = 0; i < n; ++i) s += hp[(size_t)i * DD];
  pooled[(size_t)b * DD + col] = s / cnt + hp[0];  // + h[target]
}

// ---------------------------------------------------------------- L2 norm
__global__ void l2norm_kernel(const float* __restrict__ emb, float* __restrict__ outp) {
  __shared__ float red[DD];
  int b = blockIdx.x, c = threadIdx.x;
  float v = emb[(size_t)b * DD + c];
  red[c] = v * v;
  __syncthreads();
  for (int s = 64; s > 0; s >>= 1) {
    if (c < s) red[c] += red[c + s];
    __syncthreads();
  }
  float nrm = fmaxf(sqrtf(red[0]), 1e-12f);
  outp[(size_t)b * DD + c] = v / nrm;
}

// ----------------------------------------------------------------
extern "C" void kernel_launch(void* const* d_in, const int* in_sizes, int n_in,
                              void* d_out, int out_size, void* d_ws, size_t ws_size,
                              hipStream_t stream) {
  const float* feat      = (const float*)d_in[0];
  const float* edge_w    = (const float*)d_in[1];
  const float* W1        = (const float*)d_in[2];
  const float* b1        = (const float*)d_in[3];
  const float* W2        = (const float*)d_in[4];
  const float* b2        = (const float*)d_in[5];
  const float* Wp        = (const float*)d_in[6];
  const float* bp        = (const float*)d_in[7];
  const float* size_subg = (const float*)d_in[8];
  const int*   edge_src  = (const int*)d_in[9];
  const int*   edge_dst  = (const int*)d_in[10];
  const int*   target    = (const int*)d_in[12];

  const int N = in_sizes[0] / DD;
  const int E = in_sizes[9];
  const int B = in_sizes[12];

  // workspace carve-out
  char*  ws  = (char*)d_ws;
  size_t off = 0;
  float* agg = (float*)(ws + off); off += (size_t)N * DD * sizeof(float);
  float* h   = (float*)(ws + off); off += (size_t)N * DD * sizeof(float);
  bf16*  Wf  = (bf16*) (ws + off); off += (size_t)3 * 16384 * sizeof(bf16);
  off = (off + 255) & ~(size_t)255;
  float* pooled = (float*)(ws + off); off += (size_t)B * DD * sizeof(float);
  off = (off + 255) & ~(size_t)255;
  float* emb    = (float*)(ws + off);

  bf16* W1f = Wf;
  bf16* W2f = Wf + 16384;
  bf16* Wpf = Wf + 32768;

  const int n4        = (N * DD) / 4;
  const int zeroBlks  = (n4 + 255) / 256;
  const int spmmBlks  = (E + 7) / 8;            // 8 waves (edges) per block
  const int gemmBlks  = (N + 15) / 16;          // N is a multiple of 16
  const int poolGemmB = (B + 15) / 16;

  // pack weights into WMMA bf16 fragments
  prep_wfrag_kernel<<<192, 256, 0, stream>>>(W1, W2, Wp, Wf);

  // layer 1: agg = A_hat @ feat ; h = relu(agg @ W1 + b1)  (h holds h1)
  zero_f32_kernel<<<zeroBlks, 256, 0, stream>>>(agg, n4);
  spmm_kernel<<<spmmBlks, 256, 0, stream>>>(feat, edge_w, edge_src, edge_dst, agg, E);
  gcn_gemm_kernel<false, false><<<gemmBlks, 256, 0, stream>>>(agg, W1f, b1, nullptr, h, N);

  // layer 2: agg = A_hat @ h1 ; h = h1 + relu(agg @ W2 + b2)  (fused residual)
  zero_f32_kernel<<<zeroBlks, 256, 0, stream>>>(agg, n4);
  spmm_kernel<<<spmmBlks, 256, 0, stream>>>(h, edge_w, edge_src, edge_dst, agg, E);
  gcn_gemm_kernel<true, false><<<gemmBlks, 256, 0, stream>>>(agg, W2f, b2, h, h, N);

  // pool + projection + normalize
  pool_kernel<<<B, DD, 0, stream>>>(h, target, size_subg, pooled);
  gcn_gemm_kernel<false, true><<<poolGemmB, 256, 0, stream>>>(pooled, Wpf, bp, nullptr, emb, B);
  l2norm_kernel<<<B, DD, 0, stream>>>(emb, (float*)d_out);
}
